// my_conv_3762391352099
// MI455X (gfx1250) — matrix operand order repaired and checked by reading, not compiled, not run on hardware
//
#include <hip/hip_runtime.h>

typedef __attribute__((ext_vector_type(16))) _Float16 v16h;
typedef __attribute__((ext_vector_type(8)))  _Float16 v8h;
typedef __attribute__((ext_vector_type(2)))  _Float16 v2h;
typedef __attribute__((ext_vector_type(8)))  float    v8f;

#define IN_CH  3
#define OUT_CH 16
#define IW     256
#define IH     256
#define OWW    254
#define OHH    254
#define KK     27      // IN_CH * 3 * 3 reduction length
#define KPAD   32      // padded to WMMA K

// One 512-thread block (16 wave32s) per (n, oh) output row.
// Block stages (a) f16 zero-padded weights wh[oc][32] and (b) a f16 im2col
// tile imcol[px][k] (px = output column 0..255, k = ic*9+kh*3+kw padded to 32)
// in LDS. Each wave then builds its WMMA operands with two ds_load_b128 each
// and computes a 16(oc) x 16(ow) tile with ONE v_wmma_f32_16x16x32_f16:
//   A (16x32 f16) = weights  (M = oc)          lanes 0-15: K{0..7,16..23}
//                                              lanes 16-31: K{8..15,24..31}
//   B (32x16 f16) = im2col   (N = ow in tile)  lanes 0-15: K{0..7,16..23}
//                                              lanes 16-31: K{8..15,24..31}
//   C/D (16x16 f32): M = oc, N = ow  -> 64B coalesced stores per half-wave.
__global__ __launch_bounds__(512)
void conv3x3_wmma_kernel(const float* __restrict__ x,
                         const float* __restrict__ w,
                         const float* __restrict__ bias,
                         float* __restrict__ out)
{
    __shared__ __align__(16) _Float16 imcol[IW * KPAD];     // 16 KB, [px][k]
    __shared__ __align__(16) _Float16 wh[OUT_CH * KPAD];    //  1 KB, [oc][k]
    __shared__ float bs[OUT_CH];

    const int tid = threadIdx.x;
    const int n   = blockIdx.x / OHH;
    const int oh  = blockIdx.x % OHH;

    // ---- Stage weights as f16, zero-padded in K (A-operand friendly layout).
    if (tid < OUT_CH * KPAD / 2) {
        int oc = tid >> 4;
        int k0 = (tid & 15) * 2, k1 = k0 + 1;
        float f0 = (k0 < KK) ? w[oc * KK + k0] : 0.0f;
        float f1 = (k1 < KK) ? w[oc * KK + k1] : 0.0f;
        *(v2h*)(wh + oc * KPAD + k0) = (v2h){(_Float16)f0, (_Float16)f1};
    }
    if (tid < OUT_CH) bs[tid] = bias[tid];

    // ---- Stage f16 im2col for this output row: imcol[px][k].
    // i = p*256 + px  (p = k-pair index): consecutive tids share p and walk px,
    // so the two global loads per iteration are coalesced row reads (L2 hits;
    // whole x fits in the 192MB L2, so HBM sees x roughly once).
    const float* xb = x + ((size_t)n * IN_CH * IH + oh) * IW;
    #pragma unroll
    for (int it = 0; it < (IW * KPAD / 2) / 512; ++it) {   // 8 iterations
        int i  = tid + it * 512;
        int px = i & (IW - 1);
        int p  = i >> 8;                   // k-pair 0..15
        int k0 = 2 * p, k1 = k0 + 1;
        float f0 = 0.0f, f1 = 0.0f;
        if (k0 < KK) {
            int r = k0 / 3, cc = k0 - 3 * r;       // r = ic*3 + kh
            int ic = r / 3, kh = r - 3 * ic;
            int col = px + cc;                     // <= 257: guard row edge
            if (col < IW) f0 = xb[((size_t)ic * IH + kh) * IW + col];
        }
        if (k1 < KK) {
            int r = k1 / 3, cc = k1 - 3 * r;
            int ic = r / 3, kh = r - 3 * ic;
            int col = px + cc;
            if (col < IW) f1 = xb[((size_t)ic * IH + kh) * IW + col];
        }
        *(v2h*)(imcol + px * KPAD + k0) = (v2h){(_Float16)f0, (_Float16)f1};
    }
    __syncthreads();

    const int  wave = tid >> 5;          // 0..15 -> ow tile
    const int  lane = tid & 31;
    const int  half = lane & 15;
    const bool up   = lane >= 16;
    const int  ow0  = wave * 16;

    // ---- A fragment: two 16B-aligned ds_load_b128 from wh.
    const int ka = up ? 8 : 0;
    v8h a0 = *(const v8h*)(wh + half * KPAD + ka);
    v8h a1 = *(const v8h*)(wh + half * KPAD + ka + 16);
    v16h a = __builtin_shufflevector(a0, a1, 0, 1, 2, 3, 4, 5, 6, 7,
                                             8, 9, 10, 11, 12, 13, 14, 15);

    // ---- B fragment: two 16B-aligned ds_load_b128 from imcol.
    const int px = ow0 + half;
    const int kb = up ? 8 : 0;
    v8h b0 = *(const v8h*)(imcol + px * KPAD + kb);
    v8h b1 = *(const v8h*)(imcol + px * KPAD + kb + 16);
    v16h b = __builtin_shufflevector(b0, b1, 0, 1, 2, 3, 4, 5, 6, 7,
                                             8, 9, 10, 11, 12, 13, 14, 15);

    // EXEC is all-1s here (no live divergence): required for WMMA.
    v8f c = {};
    c = __builtin_amdgcn_wmma_f32_16x16x32_f16(
            /*neg_a=*/false, a, /*neg_b=*/false, b,
            /*c_mod=*/(short)0, c, /*reuse_a=*/false, /*reuse_b=*/false);

    // D layout: VGPR v, lanes 0-15 -> (M=v, N=lane); lanes 16-31 -> (M=v+8).
    // M = oc, N = ow offset -> each half-wave store is 64B contiguous in ow.
    const int ow = ow0 + half;
    if (ow < OWW) {
        #pragma unroll
        for (int v = 0; v < 8; ++v) {
            int oc = v + (up ? 8 : 0);
            out[((size_t)(n * OUT_CH + oc) * OHH + oh) * OWW + ow] = c[v] + bs[oc];
        }
    }
}

extern "C" void kernel_launch(void* const* d_in, const int* in_sizes, int n_in,
                              void* d_out, int out_size, void* d_ws, size_t ws_size,
                              hipStream_t stream) {
    (void)n_in; (void)out_size; (void)d_ws; (void)ws_size;
    const float* x    = (const float*)d_in[0];
    const float* w    = (const float*)d_in[1];
    const float* bias = (const float*)d_in[2];
    float*       out  = (float*)d_out;

    const int nbatch = in_sizes[0] / (IN_CH * IH * IW);   // 64
    dim3 grid((unsigned)(nbatch * OHH));                  // one block per (n, oh)
    conv3x3_wmma_kernel<<<grid, 512, 0, stream>>>(x, w, bias, out);
}
